// StatefulQwen2Decoder_60095182405736
// MI455X (gfx1250) — matrix operand order repaired and checked by reading, not compiled, not run on hardware
//
#include <hip/hip_runtime.h>
#include <hip/hip_bf16.h>
#include <math.h>

typedef _Float16 half_t;
typedef __attribute__((ext_vector_type(16))) _Float16 v16h;
typedef __attribute__((ext_vector_type(8)))  float    v8f;

// ---- model constants ----
#define LAYERS 20
#define QLEN   512
#define PASTL  512
#define ENDL   1024
#define MAXS   2048
#define DMODEL 896
#define NKV    2
#define NQ     14
#define GRP    7
#define HD     64
#define FF     4864
#define EPSV   1e-6f
#define SCALEV 0.125f   // 1/sqrt(64)

union Frag16 {            // 16 halves per lane = one 16x32 A or 32x16 B fragment
    v16h v;
    uint4 q[2];
};

__device__ __forceinline__ v8f zero8() {
    v8f r;
#pragma unroll
    for (int i = 0; i < 8; ++i) r[i] = 0.0f;
    return r;
}

__device__ __forceinline__ v8f wmma16(const Frag16& a, const Frag16& b, v8f c) {
    return __builtin_amdgcn_wmma_f32_16x16x32_f16(false, a.v, false, b.v,
                                                  (short)0, c, false, false);
}

__device__ __forceinline__ float rmax16(float x) {
#pragma unroll
    for (int m = 1; m < 16; m <<= 1) x = fmaxf(x, __shfl_xor(x, m, 16));
    return x;
}
__device__ __forceinline__ float rsum16(float x) {
#pragma unroll
    for (int m = 1; m < 16; m <<= 1) x += __shfl_xor(x, m, 16);
    return x;
}

// ---- CDNA5 async global->LDS copy (ASYNCcnt-tracked, bypasses VGPRs) ----
// ISA: GLOBAL_LOAD_ASYNC_TO_LDS_B128  (VDST = LDS byte address VGPR, VADDR = 64b global)
__device__ __forceinline__ void async_copy_b128(void* lds_dst, const void* gsrc) {
    unsigned lds_off = (unsigned)(unsigned long long)lds_dst;  // low 32b of generic = LDS addr
    asm volatile("global_load_async_to_lds_b128 %0, %1, off"
                 :
                 : "v"(lds_off), "v"(gsrc)
                 : "memory");
}
__device__ __forceinline__ void wait_async0() {
#if __has_builtin(__builtin_amdgcn_s_wait_asynccnt)
    __builtin_amdgcn_s_wait_asynccnt(0);
#else
    asm volatile("s_wait_asynccnt 0x0" ::: "memory");
#endif
}

// =====================================================================
// RMSNorm: one block per row (512 rows, 896 cols). F16OUT -> half output
// (for WMMA A operands), else float (final norm -> d_out).
// =====================================================================
template <bool F16OUT>
__global__ __launch_bounds__(256) void rmsnorm_kernel(const float* __restrict__ x,
                                                      const float* __restrict__ w,
                                                      void* __restrict__ out) {
    const int row = blockIdx.x;
    const int tid = threadIdx.x;
    const int lane = tid & 31, wave = tid >> 5;
    const float* xr = x + (size_t)row * DMODEL;

    float s = 0.0f;
    for (int i = tid; i < DMODEL; i += 256) {
        float v = xr[i];
        s += v * v;
    }
#pragma unroll
    for (int m = 1; m < 32; m <<= 1) s += __shfl_xor(s, m, 32);

    __shared__ float red[8];
    __shared__ float scale;
    if (lane == 0) red[wave] = s;
    __syncthreads();
    if (tid == 0) {
        float t = 0.0f;
#pragma unroll
        for (int i = 0; i < 8; ++i) t += red[i];
        scale = rsqrtf(t / (float)DMODEL + EPSV);
    }
    __syncthreads();
    const float sc = scale;
    for (int i = tid; i < DMODEL; i += 256) {
        float v = xr[i] * sc * w[i];
        if (F16OUT)
            ((half_t*)out)[(size_t)row * DMODEL + i] = (half_t)v;
        else
            ((float*)out)[(size_t)row * DMODEL + i] = v;
    }
}

// =====================================================================
// WMMA GEMM: C[M,N] = A_f16[M,K] @ B_f32->f16[K,N]  (+ epilogue)
//   MODE 0: out float = C + bias[n]
//   MODE 1: out float = C + resid[m,n]        (resid may alias out)
//   MODE 2: two B matrices (gate/up); out half = silu(Cg) * Cu
// Block: 256 threads = 8 waves, tile 128(M) x 64(N), K-step 32.
// Double-buffered LDS; A tiles staged with global_load_async_to_lds_b128
// (ASYNCcnt), B tiles stream fp32->f16 through VGPRs overlapped with WMMA.
// =====================================================================
#define BM 128
#define BN 64
#define BK 32

template <int MODE>
__global__ __launch_bounds__(256) void gemm_wmma(const half_t* __restrict__ A,
                                                 const float* __restrict__ B0,
                                                 const float* __restrict__ B1,
                                                 const float* __restrict__ bias,
                                                 const float* __restrict__ resid,
                                                 float* __restrict__ outF,
                                                 half_t* __restrict__ outH,
                                                 int M, int N, int K) {
    __shared__ __align__(16) half_t As[2][BM][BK];
    __shared__ __align__(16) half_t Bt[2][BN][BK];    // [n][k] so B frags are contiguous
    __shared__ __align__(16) half_t Bt2[2][BN][BK];

    const int tid  = threadIdx.x;
    const int lane = tid & 31;
    const int wave = tid >> 5;
    const int bn0  = blockIdx.x * BN;
    const int bm0  = blockIdx.y * BM;
    const int wm   = (wave >> 1) * 32;   // 4 waves along M
    const int wn   = (wave & 1) * 32;    // 2 waves along N
    const int lrow = lane & 15;
    const int hi   = (lane & 16) ? 1 : 0;

    // staging decomposition (fixed per thread)
    const int ar0 = tid >> 2,            ac0 = (tid & 3) * 8;          // A piece 0
    const int ar1 = (tid + 256) >> 2,    ac1 = ((tid + 256) & 3) * 8;  // A piece 1
    const int kr0 = tid >> 4,            nc0 = (tid & 15) * 4;         // B piece 0
    const int kr1 = (tid + 256) >> 4,    nc1 = ((tid + 256) & 15) * 4; // B piece 1

    v8f acc[2][2], acc2[2][2];
#pragma unroll
    for (int i = 0; i < 2; ++i)
#pragma unroll
        for (int j = 0; j < 2; ++j) { acc[i][j] = zero8(); acc2[i][j] = zero8(); }

    float4 breg[2], breg2[2];

    auto stageA = [&](int k0, int buf) {
        async_copy_b128(&As[buf][ar0][ac0], &A[(size_t)(bm0 + ar0) * K + k0 + ac0]);
        async_copy_b128(&As[buf][ar1][ac1], &A[(size_t)(bm0 + ar1) * K + k0 + ac1]);
    };
    auto loadB = [&](int k0) {
        breg[0] = *reinterpret_cast<const float4*>(&B0[(size_t)(k0 + kr0) * N + bn0 + nc0]);
        breg[1] = *reinterpret_cast<const float4*>(&B0[(size_t)(k0 + kr1) * N + bn0 + nc1]);
        if (MODE == 2) {
            breg2[0] = *reinterpret_cast<const float4*>(&B1[(size_t)(k0 + kr0) * N + bn0 + nc0]);
            breg2[1] = *reinterpret_cast<const float4*>(&B1[(size_t)(k0 + kr1) * N + bn0 + nc1]);
        }
    };
    auto storeB = [&](int buf) {
        Bt[buf][nc0 + 0][kr0] = (half_t)breg[0].x;
        Bt[buf][nc0 + 1][kr0] = (half_t)breg[0].y;
        Bt[buf][nc0 + 2][kr0] = (half_t)breg[0].z;
        Bt[buf][nc0 + 3][kr0] = (half_t)breg[0].w;
        Bt[buf][nc1 + 0][kr1] = (half_t)breg[1].x;
        Bt[buf][nc1 + 1][kr1] = (half_t)breg[1].y;
        Bt[buf][nc1 + 2][kr1] = (half_t)breg[1].z;
        Bt[buf][nc1 + 3][kr1] = (half_t)breg[1].w;
        if (MODE == 2) {
            Bt2[buf][nc0 + 0][kr0] = (half_t)breg2[0].x;
            Bt2[buf][nc0 + 1][kr0] = (half_t)breg2[0].y;
            Bt2[buf][nc0 + 2][kr0] = (half_t)breg2[0].z;
            Bt2[buf][nc0 + 3][kr0] = (half_t)breg2[0].w;
            Bt2[buf][nc1 + 0][kr1] = (half_t)breg2[1].x;
            Bt2[buf][nc1 + 1][kr1] = (half_t)breg2[1].y;
            Bt2[buf][nc1 + 2][kr1] = (half_t)breg2[1].z;
            Bt2[buf][nc1 + 3][kr1] = (half_t)breg2[1].w;
        }
    };

    const int T = K / BK;
    // prologue: tile 0
    stageA(0, 0);
    loadB(0);
    storeB(0);

    for (int t = 0; t < T; ++t) {
        const int b = t & 1;
        wait_async0();        // this wave's async A copies for buffer b are complete
        __syncthreads();      // all waves' A + B stores for buffer b visible; prev reads of b^1 done

        if (t + 1 < T) {
            stageA((t + 1) * BK, b ^ 1);   // async stream next A tile into the idle buffer
            loadB((t + 1) * BK);           // next B tile -> VGPRs, latency hidden by WMMAs
        }

        // ---- fragments (ISA 16-bit A/B VGPR layouts) ----
        Frag16 a[2], bfr[2], b2fr[2];
        const int ac = hi ? 8 : 0;    // A: lo lanes K0..7/K16..23, hi lanes K8..15/K24..31
#pragma unroll
        for (int mi = 0; mi < 2; ++mi) {
            const half_t* ar = &As[b][wm + mi * 16 + lrow][0];
            a[mi].q[0] = *reinterpret_cast<const uint4*>(ar + ac);
            a[mi].q[1] = *reinterpret_cast<const uint4*>(ar + ac + 16);
        }
        const int kb = hi ? 8 : 0;    // B: lo lanes K0..7/K16..23, hi lanes K8..15/K24..31
#pragma unroll
        for (int ni = 0; ni < 2; ++ni) {
            const half_t* br = &Bt[b][wn + ni * 16 + lrow][0];
            bfr[ni].q[0] = *reinterpret_cast<const uint4*>(br + kb);
            bfr[ni].q[1] = *reinterpret_cast<const uint4*>(br + kb + 16);
            if (MODE == 2) {
                const half_t* br2 = &Bt2[b][wn + ni * 16 + lrow][0];
                b2fr[ni].q[0] = *reinterpret_cast<const uint4*>(br2 + kb);
                b2fr[ni].q[1] = *reinterpret_cast<const uint4*>(br2 + kb + 16);
            }
        }
#pragma unroll
        for (int mi = 0; mi < 2; ++mi)
#pragma unroll
            for (int ni = 0; ni < 2; ++ni) {
                acc[mi][ni] = wmma16(a[mi], bfr[ni], acc[mi][ni]);
                if (MODE == 2) acc2[mi][ni] = wmma16(a[mi], b2fr[ni], acc2[mi][ni]);
            }

        if (t + 1 < T) storeB(b ^ 1);  // convert+store next B tile after this tile's reads
    }

    // ---- epilogue (C layout: VGPR i -> row i (lo) / i+8 (hi); col = lane&15) ----
#pragma unroll
    for (int mi = 0; mi < 2; ++mi)
#pragma unroll
        for (int ni = 0; ni < 2; ++ni)
#pragma unroll
            for (int i = 0; i < 8; ++i) {
                int m = bm0 + wm + mi * 16 + i + hi * 8;
                int n = bn0 + wn + ni * 16 + lrow;
                float v = acc[mi][ni][i];
                if (MODE == 0) {
                    outF[(size_t)m * N + n] = v + bias[n];
                } else if (MODE == 1) {
                    outF[(size_t)m * N + n] = v + resid[(size_t)m * N + n];
                } else {
                    float g = v;
                    float u = acc2[mi][ni][i];
                    float s = g / (1.0f + __expf(-g));
                    outH[(size_t)m * N + n] = (half_t)(s * u);
                }
            }
}

// =====================================================================
// RoPE on Q heads (896 = 14*64 cols), write f16 activation for attention.
// =====================================================================
__global__ __launch_bounds__(256) void rope_q_kernel(const float* __restrict__ qf,
                                                     const float* __restrict__ pcos,
                                                     const float* __restrict__ psin,
                                                     half_t* __restrict__ qh) {
    int idx = blockIdx.x * 256 + threadIdx.x;
    if (idx >= QLEN * DMODEL) return;
    int q = idx / DMODEL;
    int col = idx - q * DMODEL;
    int d = col & 63;
    float v = qf[(size_t)q * DMODEL + col];
    float p = (d < 32) ? -qf[(size_t)q * DMODEL + col + 32]
                       :  qf[(size_t)q * DMODEL + col - 32];
    float r = v * pcos[q * HD + d] + p * psin[q * HD + d];
    qh[(size_t)q * DMODEL + col] = (half_t)r;
}

// =====================================================================
// RoPE + inject blend on K, inject blend on V; write new tokens into
// k_all (row-major [g][key][d]) and vt_all (d-major [g][d][key]).
// =====================================================================
__global__ __launch_bounds__(256) void rope_kv_kernel(const float* __restrict__ kf,
                                                      const float* __restrict__ vf,
                                                      const float* __restrict__ pcos,
                                                      const float* __restrict__ psin,
                                                      const float* __restrict__ injm,
                                                      const float* __restrict__ ik,
                                                      const float* __restrict__ iv,
                                                      half_t* __restrict__ kall,
                                                      half_t* __restrict__ vt) {
    int idx = blockIdx.x * 256 + threadIdx.x;
    if (idx >= QLEN * NKV * HD) return;
    int q = idx / (NKV * HD);
    int rem = idx - q * (NKV * HD);
    int g = rem >> 6;
    int d = rem & 63;
    float inj = injm[0];
    float nrm = 1.0f - inj;

    float kv = kf[(size_t)q * (NKV * HD) + rem];
    float kp = (d < 32) ? -kf[(size_t)q * (NKV * HD) + rem + 32]
                        :  kf[(size_t)q * (NKV * HD) + rem - 32];
    float kr = kv * pcos[q * HD + d] + kp * psin[q * HD + d];
    float kfin = nrm * kr + inj * ik[((size_t)g * QLEN + q) * HD + d];
    kall[((size_t)g * ENDL + PASTL + q) * HD + d] = (half_t)kfin;

    float vv = vf[(size_t)q * (NKV * HD) + rem];
    float vfin = nrm * vv + inj * iv[((size_t)g * QLEN + q) * HD + d];
    vt[((size_t)g * HD + d) * ENDL + PASTL + q] = (half_t)vfin;
}

// Copy the cached first PAST tokens into k_all / vt_all (with transpose for V).
__global__ __launch_bounds__(256) void cache_copy_kernel(const half_t* __restrict__ kc,
                                                         const half_t* __restrict__ vc,
                                                         half_t* __restrict__ kall,
                                                         half_t* __restrict__ vt) {
    int idx = blockIdx.x * 256 + threadIdx.x;
    if (idx >= NKV * PASTL * HD) return;
    int g = idx / (PASTL * HD);
    int rem = idx - g * (PASTL * HD);
    int kk = rem >> 6;
    int d = rem & 63;
    size_t src = ((size_t)g * MAXS + kk) * HD + d;
    kall[((size_t)g * ENDL + kk) * HD + d] = kc[src];
    vt[((size_t)g * HD + d) * ENDL + kk] = vc[src];
}

// =====================================================================
// Flash attention: one wave per (head, 16-query tile). 14*32 = 448 blocks.
// S = Q(16x64) @ K^T via 2 chained WMMAs per 16-key tile; online softmax
// with shfl-based row reductions; P staged through LDS (C->A relayout);
// O += P(16x32) @ V(32x64) via 4 WMMAs per 32-key chunk (Vt is d-major).
// =====================================================================
__global__ __launch_bounds__(32) void attn_wmma_kernel(const half_t* __restrict__ qh,
                                                       const half_t* __restrict__ kall,
                                                       const half_t* __restrict__ vt,
                                                       const float* __restrict__ mask,
                                                       half_t* __restrict__ oh) {
    __shared__ __align__(16) half_t Pbuf[16][32];

    const int lane = threadIdx.x & 31;
    const int h  = blockIdx.x >> 5;    // 0..13
    const int qt = blockIdx.x & 31;    // 0..31
    const int g  = h / GRP;
    const int q0 = qt * 16;
    const int lrow = lane & 15;
    const int hi = (lane & 16) ? 1 : 0;

    // Q fragments: two 16x32 A-frags covering d=0..31 and d=32..63
    Frag16 aq[2];
#pragma unroll
    for (int f = 0; f < 2; ++f) {
        const half_t* qr = &qh[(size_t)(q0 + lrow) * DMODEL + h * HD + f * 32 + (hi ? 8 : 0)];
        aq[f].q[0] = *reinterpret_cast<const uint4*>(qr);
        aq[f].q[1] = *reinterpret_cast<const uint4*>(qr + 16);
    }

    float m8[8], l8[8];
#pragma unroll
    for (int i = 0; i < 8; ++i) { m8[i] = -1e30f; l8[i] = 0.0f; }
    v8f o[4];
#pragma unroll
    for (int j = 0; j < 4; ++j) o[j] = zero8();

    for (int kt = 0; kt < ENDL / 32; ++kt) {
        const int k0 = kt * 32;
        v8f s[2];
#pragma unroll
        for (int nt = 0; nt < 2; ++nt) {
            s[nt] = zero8();
            const int key = k0 + nt * 16 + lrow;
            const half_t* krow = &kall[((size_t)g * ENDL + key) * HD];
#pragma unroll
            for (int f = 0; f < 2; ++f) {
                Frag16 bk;
                const int db = f * 32 + (hi ? 8 : 0);
                bk.q[0] = *reinterpret_cast<const uint4*>(krow + db);
                bk.q[1] = *reinterpret_cast<const uint4*>(krow + db + 16);
                s[nt] = wmma16(aq[f], bk, s[nt]);
            }
#pragma unroll
            for (int i = 0; i < 8; ++i) {
                int qq = q0 + i + hi * 8;
                int kk = k0 + nt * 16 + lrow;
                s[nt][i] = s[nt][i] * SCALEV + mask[(size_t)qq * ENDL + kk];
            }
        }
        // online softmax per row (rows 0..7 in lo half-wave, 8..15 in hi)
#pragma unroll
        for (int i = 0; i < 8; ++i) {
            float mx = rmax16(fmaxf(s[0][i], s[1][i]));
            float mnew = fmaxf(m8[i], mx);
            float alpha = __expf(m8[i] - mnew);
            m8[i] = mnew;
            float p0 = __expf(s[0][i] - mnew);
            float p1 = __expf(s[1][i] - mnew);
            l8[i] = l8[i] * alpha + rsum16(p0 + p1);
#pragma unroll
            for (int j = 0; j < 4; ++j) o[j][i] *= alpha;
            Pbuf[i + hi * 8][lrow] = (half_t)p0;
            Pbuf[i + hi * 8][16 + lrow] = (half_t)p1;
        }
        __syncthreads();
        Frag16 ap;
        {
            const half_t* pr = &Pbuf[lrow][hi ? 8 : 0];
            ap.q[0] = *reinterpret_cast<const uint4*>(pr);
            ap.q[1] = *reinterpret_cast<const uint4*>(pr + 16);
        }
#pragma unroll
        for (int j = 0; j < 4; ++j) {
            Frag16 bv;
            const int d = j * 16 + lrow;
            const half_t* vr = &vt[((size_t)g * HD + d) * ENDL + k0 + (hi ? 8 : 0)];
            bv.q[0] = *reinterpret_cast<const uint4*>(vr);
            bv.q[1] = *reinterpret_cast<const uint4*>(vr + 16);
            o[j] = wmma16(ap, bv, o[j]);
        }
        __syncthreads();
    }
    // normalize + store f16 (row-major [q][NQ*HD] so Wo GEMM can consume as A)
#pragma unroll
    for (int i = 0; i < 8; ++i) {
        float rl = 1.0f / l8[i];
        int qq = q0 + i + hi * 8;
#pragma unroll
        for (int j = 0; j < 4; ++j)
            oh[(size_t)qq * DMODEL + h * HD + j * 16 + lrow] = (half_t)(o[j][i] * rl);
    }
}

// =====================================================================
// Host-side orchestration
// =====================================================================
extern "C" void kernel_launch(void* const* d_in, const int* in_sizes, int n_in,
                              void* d_out, int out_size, void* d_ws, size_t ws_size,
                              hipStream_t stream) {
    const float*  hidden  = (const float*)d_in[0];
    const float*  pcos    = (const float*)d_in[1];
    const float*  psin    = (const float*)d_in[2];
    const float*  mask    = (const float*)d_in[3];
    const float*  injm    = (const float*)d_in[4];
    const float*  ik      = (const float*)d_in[5];
    const float*  iv      = (const float*)d_in[6];
    const half_t* kcache  = (const half_t*)d_in[7];
    const half_t* vcache  = (const half_t*)d_in[8];
    const float*  Wq      = (const float*)d_in[9];
    const float*  bq      = (const float*)d_in[10];
    const float*  Wk      = (const float*)d_in[11];
    const float*  bk      = (const float*)d_in[12];
    const float*  Wv      = (const float*)d_in[13];
    const float*  bv      = (const float*)d_in[14];
    const float*  Wo      = (const float*)d_in[15];
    const float*  ln1     = (const float*)d_in[16];
    const float*  ln2     = (const float*)d_in[17];
    const float*  Wg      = (const float*)d_in[18];
    const float*  Wu      = (const float*)d_in[19];
    const float*  Wd      = (const float*)d_in[20];
    const float*  finalw  = (const float*)d_in[21];

    // ---- workspace carve-out (256B aligned regions) ----
    char* wsb = (char*)d_ws;
    size_t off = 0;
    auto carve = [&](size_t bytes) -> void* {
        void* p = wsb + off;
        off += (bytes + 255) & ~(size_t)255;
        return p;
    };
    float*  hbuf = (float*)carve((size_t)QLEN * DMODEL * 4);
    half_t* xh   = (half_t*)carve((size_t)QLEN * DMODEL * 2);
    float*  qf   = (float*)carve((size_t)QLEN * DMODEL * 4);
    float*  kf   = (float*)carve((size_t)QLEN * NKV * HD * 4);
    float*  vf   = (float*)carve((size_t)QLEN * NKV * HD * 4);
    half_t* qh   = (half_t*)carve((size_t)QLEN * DMODEL * 2);
    half_t* kall = (half_t*)carve((size_t)NKV * ENDL * HD * 2);
    half_t* vt   = (half_t*)carve((size_t)NKV * HD * ENDL * 2);
    half_t* oh   = (half_t*)carve((size_t)QLEN * DMODEL * 2);
    half_t* x2h  = (half_t*)carve((size_t)QLEN * DMODEL * 2);
    half_t* act  = (half_t*)carve((size_t)QLEN * FF * 2);
    (void)ws_size; (void)n_in; (void)in_sizes; (void)out_size;

    hipMemcpyAsync(hbuf, hidden, (size_t)QLEN * DMODEL * 4,
                   hipMemcpyDeviceToDevice, stream);

    const dim3 blk256(256), blk32(32);
    for (int l = 0; l < LAYERS; ++l) {
        const float* Wq_l = Wq + (size_t)l * DMODEL * DMODEL;
        const float* Wk_l = Wk + (size_t)l * DMODEL * NKV * HD;
        const float* Wv_l = Wv + (size_t)l * DMODEL * NKV * HD;
        const float* Wo_l = Wo + (size_t)l * DMODEL * DMODEL;
        const float* Wg_l = Wg + (size_t)l * DMODEL * FF;
        const float* Wu_l = Wu + (size_t)l * DMODEL * FF;
        const float* Wd_l = Wd + (size_t)l * FF * DMODEL;
        const float* bq_l = bq + (size_t)l * DMODEL;
        const float* bk_l = bk + (size_t)l * NKV * HD;
        const float* bv_l = bv + (size_t)l * NKV * HD;
        const float* ln1_l = ln1 + (size_t)l * DMODEL;
        const float* ln2_l = ln2 + (size_t)l * DMODEL;
        const float* ik_l = ik + (size_t)l * NKV * QLEN * HD;
        const float* iv_l = iv + (size_t)l * NKV * QLEN * HD;
        const half_t* kc_l = kcache + (size_t)l * NKV * MAXS * HD;
        const half_t* vc_l = vcache + (size_t)l * NKV * MAXS * HD;

        rmsnorm_kernel<true><<<QLEN, blk256, 0, stream>>>(hbuf, ln1_l, xh);

        gemm_wmma<0><<<dim3(DMODEL / BN, QLEN / BM), blk256, 0, stream>>>(
            xh, Wq_l, nullptr, bq_l, nullptr, qf, nullptr, QLEN, DMODEL, DMODEL);
        gemm_wmma<0><<<dim3((NKV * HD) / BN, QLEN / BM), blk256, 0, stream>>>(
            xh, Wk_l, nullptr, bk_l, nullptr, kf, nullptr, QLEN, NKV * HD, DMODEL);
        gemm_wmma<0><<<dim3((NKV * HD) / BN, QLEN / BM), blk256, 0, stream>>>(
            xh, Wv_l, nullptr, bv_l, nullptr, vf, nullptr, QLEN, NKV * HD, DMODEL);

        rope_q_kernel<<<(QLEN * DMODEL) / 256, blk256, 0, stream>>>(qf, pcos, psin, qh);
        rope_kv_kernel<<<(QLEN * NKV * HD) / 256, blk256, 0, stream>>>(
            kf, vf, pcos, psin, injm, ik_l, iv_l, kall, vt);
        cache_copy_kernel<<<(NKV * PASTL * HD) / 256, blk256, 0, stream>>>(
            kc_l, vc_l, kall, vt);

        attn_wmma_kernel<<<NQ * (QLEN / 16), blk32, 0, stream>>>(qh, kall, vt, mask, oh);

        gemm_wmma<1><<<dim3(DMODEL / BN, QLEN / BM), blk256, 0, stream>>>(
            oh, Wo_l, nullptr, nullptr, hbuf, hbuf, nullptr, QLEN, DMODEL, DMODEL);

        rmsnorm_kernel<true><<<QLEN, blk256, 0, stream>>>(hbuf, ln2_l, x2h);

        gemm_wmma<2><<<dim3(FF / BN, QLEN / BM), blk256, 0, stream>>>(
            x2h, Wg_l, Wu_l, nullptr, nullptr, nullptr, act, QLEN, FF, DMODEL);

        gemm_wmma<1><<<dim3(DMODEL / BN, QLEN / BM), blk256, 0, stream>>>(
            act, Wd_l, nullptr, nullptr, hbuf, hbuf, nullptr, QLEN, DMODEL, FF);
    }

    rmsnorm_kernel<false><<<QLEN, blk256, 0, stream>>>(hbuf, finalw, (float*)d_out);
}
